// Encoder_15212774162530
// MI455X (gfx1250) — matrix operand order repaired and checked by reading, not compiled, not run on hardware
//
#include <hip/hip_runtime.h>
#include <hip/hip_bf16.h>
#include <stdint.h>

// ---------------------------------------------------------------------------
// Bidirectional relu-LSTM encoder for MI455X (gfx1250, wave32, WMMA).
// Heavy math runs on v_wmma_f32_16x16x32_bf16 (bf16 inputs, f32 accumulate).
// Roofline: ~0.38 TFLOP total, weights (2.9MB/dir) L2-resident, e converted
// once to bf16 (84MB) so per-step GEMMs stream bf16 fragments from L2.
// ---------------------------------------------------------------------------

typedef __attribute__((ext_vector_type(16))) __bf16 v16bf;
typedef __attribute__((ext_vector_type(8)))  float  v8f;

#define BSZ  128
#define TLEN 1024
#define NE   300
#define NP   300
#define NC   64
#define G4   (4 * NP)   // 1200 gate width
#define KP   320        // padded NE / NP (multiple of 32)
#define KP2  640        // padded concat dim

union FragB {
    v16bf v;
    uint4 q[2];
    unsigned short u[16];
};

__device__ __forceinline__ unsigned short f2bf(float f) {
    unsigned int x = __float_as_uint(f);
    x += 0x7FFFu + ((x >> 16) & 1u);   // round-to-nearest-even
    return (unsigned short)(x >> 16);
}
__device__ __forceinline__ float sigm(float x) { return 1.0f / (1.0f + __expf(-x)); }

// ---------------------------------------------------------------- init kernels
__global__ void k_zero_state(float* __restrict__ cF, unsigned short* __restrict__ hB,
                             float* __restrict__ out) {
    int idx = blockIdx.x * blockDim.x + threadIdx.x;
    if (idx < 2 * BSZ * NP) cF[idx] = 0.0f;
    if (idx < 2 * BSZ * KP) hB[idx] = 0;
    if (idx == 0) out[BSZ * NC] = 0.0f;   // loss accumulator
}

// W[600,1200] -> bf16 padded [640,1200]: rows 0..299 = x-part, 320..619 = h-part
__global__ void k_conv_w(const float* __restrict__ Wf, const float* __restrict__ Wb,
                         unsigned short* __restrict__ Wp) {
    int dir = blockIdx.y;
    const float* W = dir ? Wb : Wf;
    unsigned short* dst = Wp + (size_t)dir * KP2 * G4;
    int idx = blockIdx.x * blockDim.x + threadIdx.x;
    if (idx >= KP2 * G4) return;
    int kp = idx / G4, n = idx - kp * G4;
    float v = 0.0f;
    if (kp < NE)                     v = W[(size_t)kp * G4 + n];
    else if (kp >= KP && kp < KP + NP) v = W[(size_t)(NE + kp - KP) * G4 + n];
    dst[idx] = f2bf(v);
}

// P[600,64] -> bf16 padded [640,64]
__global__ void k_conv_p(const float* __restrict__ P, unsigned short* __restrict__ Pp) {
    int idx = blockIdx.x * blockDim.x + threadIdx.x;
    if (idx >= KP2 * NC) return;
    int kp = idx / NC, n = idx - kp * NC;
    float v = 0.0f;
    if (kp < NE)                     v = P[(size_t)kp * NC + n];
    else if (kp >= KP && kp < KP + NP) v = P[(size_t)(NE + kp - KP) * NC + n];
    Pp[idx] = f2bf(v);
}

// e[128,1024,300] f32 -> bf16 padded [128,1024,320]
__global__ void k_conv_e(const float* __restrict__ e, unsigned short* __restrict__ eB) {
    int idx = blockIdx.x * blockDim.x + threadIdx.x;   // < 128*1024*320 = 41.9M
    if (idx >= BSZ * TLEN * KP) return;
    int k = idx % KP;
    int bt = idx / KP;
    float v = (k < NE) ? e[(size_t)bt * NE + k] : 0.0f;
    eB[idx] = f2bf(v);
}

// ------------------------------------------------------------- per-step GEMM
// z[dir][128][1200] = [x_t | h] @ W + bias.  One wave per 16x16 tile.
// grid = (75 N-tiles, 2 dirs), block = 256 (8 waves = 8 M-tiles).
__global__ void k_step_gemm(const unsigned short* __restrict__ eB,
                            const unsigned short* __restrict__ hB,
                            const unsigned short* __restrict__ Wp,
                            const float* __restrict__ bias_fw,
                            const float* __restrict__ bias_bw,
                            float* __restrict__ z, int s) {
    const int dir  = blockIdx.y;
    const int t    = dir ? (TLEN - 1 - s) : s;
    const int lane = threadIdx.x & 31;
    const int wave = threadIdx.x >> 5;
    const int n0   = blockIdx.x * 16;
    const int b0   = wave * 16;
    const int half = lane >> 4;
    const int m    = lane & 15;
    const int row  = b0 + m;

    const unsigned short* Wd   = Wp + (size_t)dir * KP2 * G4;
    const unsigned short* xrow = eB + ((size_t)row * TLEN + t) * KP;
    const unsigned short* hrow = hB + ((size_t)dir * BSZ + row) * KP;

    v8f acc = {};
    FragB a, b;
    // x-part: K rows 0..319 of W
    #pragma unroll
    for (int k0 = 0; k0 < KP; k0 += 32) {
        a.q[0] = *(const uint4*)(xrow + k0 + 8 * half);        // K = k0+8*half+0..7
        a.q[1] = *(const uint4*)(xrow + k0 + 16 + 8 * half);   // K = k0+16+8*half+0..7
        #pragma unroll
        for (int i = 0; i < 16; ++i)   // col N=n0+m, K = table order as A
            b.u[i] = Wd[(size_t)(k0 + 8 * half + (i & 7) + ((i >> 3) << 4)) * G4 + n0 + m];
        acc = __builtin_amdgcn_wmma_f32_16x16x32_bf16(false, a.v, false, b.v,
                                                      (short)0, acc, false, false);
    }
    // h-part: K rows 320..639 of W
    #pragma unroll
    for (int k0 = 0; k0 < KP; k0 += 32) {
        a.q[0] = *(const uint4*)(hrow + k0 + 8 * half);
        a.q[1] = *(const uint4*)(hrow + k0 + 16 + 8 * half);
        #pragma unroll
        for (int i = 0; i < 16; ++i)
            b.u[i] = Wd[(size_t)(KP + k0 + 8 * half + (i & 7) + ((i >> 3) << 4)) * G4 + n0 + m];
        acc = __builtin_amdgcn_wmma_f32_16x16x32_bf16(false, a.v, false, b.v,
                                                      (short)0, acc, false, false);
    }

    const int n  = n0 + (lane & 15);
    const float bn = (dir ? bias_bw : bias_fw)[n];
    float* zd = z + (size_t)dir * BSZ * G4;
    const int mb = b0 + (half ? 8 : 0);
    #pragma unroll
    for (int r = 0; r < 8; ++r)
        zd[(size_t)(mb + r) * G4 + n] = acc[r] + bn;           // D: M=r+8*half, N=lane&15
}

// ----------------------------------------------------------- per-step gates
__global__ void k_step_gate(const float* __restrict__ z,
                            float* __restrict__ cF,
                            unsigned short* __restrict__ hB,
                            const long long* __restrict__ nlen, int s) {
    int idx = blockIdx.x * blockDim.x + threadIdx.x;
    if (idx >= 2 * BSZ * NP) return;
    int dir = idx / (BSZ * NP);
    int r   = idx - dir * (BSZ * NP);
    int bb  = r / NP;
    int u   = r - bb * NP;
    int t   = dir ? (TLEN - 1 - s) : s;
    if ((long long)t >= nlen[bb]) return;   // freeze state past sequence length

    const float* zr = z + ((size_t)dir * BSZ + bb) * G4;
    float gi = zr[u], gj = zr[NP + u], gf = zr[2 * NP + u], go = zr[3 * NP + u];
    size_t si = (size_t)dir * BSZ * NP + r;
    float c  = cF[si];
    float cn = c * sigm(gf + 1.0f) + sigm(gi) * fmaxf(gj, 0.0f);
    float hn = fmaxf(cn, 0.0f) * sigm(go);
    cF[si] = cn;
    hB[((size_t)dir * BSZ + bb) * KP + u] = f2bf(hn);
}

// -------------------------------------------------------- projection + loss
// s = [h_fw | h_bw] @ P ; j = mean((s - s_target)^2). 1 block, 8 waves.
__global__ void k_final(const unsigned short* __restrict__ hB,
                        const unsigned short* __restrict__ Pp,
                        const float* __restrict__ sTgt,
                        float* __restrict__ out) {
    const int lane = threadIdx.x & 31;
    const int wave = threadIdx.x >> 5;      // M-tile
    const int half = lane >> 4;
    const int m    = lane & 15;
    const int row  = wave * 16 + m;

    const unsigned short* fw = hB + (size_t)row * KP;
    const unsigned short* bw = hB + ((size_t)(BSZ + row)) * KP;

    float lsum = 0.0f;
    for (int nt = 0; nt < 4; ++nt) {
        const int n0 = nt * 16;
        v8f acc = {};
        FragB a, b;
        #pragma unroll
        for (int k0 = 0; k0 < KP; k0 += 32) {
            a.q[0] = *(const uint4*)(fw + k0 + 8 * half);
            a.q[1] = *(const uint4*)(fw + k0 + 16 + 8 * half);
            #pragma unroll
            for (int i = 0; i < 16; ++i)
                b.u[i] = Pp[(size_t)(k0 + 8 * half + (i & 7) + ((i >> 3) << 4)) * NC + n0 + m];
            acc = __builtin_amdgcn_wmma_f32_16x16x32_bf16(false, a.v, false, b.v,
                                                          (short)0, acc, false, false);
        }
        #pragma unroll
        for (int k0 = 0; k0 < KP; k0 += 32) {
            a.q[0] = *(const uint4*)(bw + k0 + 8 * half);
            a.q[1] = *(const uint4*)(bw + k0 + 16 + 8 * half);
            #pragma unroll
            for (int i = 0; i < 16; ++i)
                b.u[i] = Pp[(size_t)(KP + k0 + 8 * half + (i & 7) + ((i >> 3) << 4)) * NC + n0 + m];
            acc = __builtin_amdgcn_wmma_f32_16x16x32_bf16(false, a.v, false, b.v,
                                                          (short)0, acc, false, false);
        }
        const int n  = n0 + (lane & 15);
        const int mb = wave * 16 + (half ? 8 : 0);
        #pragma unroll
        for (int r = 0; r < 8; ++r) {
            float sv = acc[r];
            int bidx = mb + r;
            out[(size_t)bidx * NC + n] = sv;
            float d = sv - sTgt[(size_t)bidx * NC + n];
            lsum += d * d;
        }
    }
    atomicAdd(out + BSZ * NC, lsum * (1.0f / (BSZ * NC)));
}

// ---------------------------------------------------------------------------
extern "C" void kernel_launch(void* const* d_in, const int* in_sizes, int n_in,
                              void* d_out, int out_size, void* d_ws, size_t ws_size,
                              hipStream_t stream) {
    const float*     e    = (const float*)d_in[0];
    const long long* nlen = (const long long*)d_in[1];   // jnp.int64
    const float*     sT   = (const float*)d_in[2];
    const float*     W_fw = (const float*)d_in[3];
    const float*     b_fw = (const float*)d_in[4];
    const float*     W_bw = (const float*)d_in[5];
    const float*     b_bw = (const float*)d_in[6];
    const float*     P    = (const float*)d_in[7];
    float*           out  = (float*)d_out;               // [128*64 s] + [1 loss]

    char*  ws  = (char*)d_ws;
    size_t off = 0;
    auto carve = [&](size_t bytes) -> char* {
        char* p = ws + off;
        off += (bytes + 255) & ~(size_t)255;
        return p;
    };
    unsigned short* eB = (unsigned short*)carve((size_t)BSZ * TLEN * KP * 2); // 83.9 MB
    unsigned short* Wp = (unsigned short*)carve((size_t)2 * KP2 * G4 * 2);    //  3.1 MB
    unsigned short* Pp = (unsigned short*)carve((size_t)KP2 * NC * 2);        //  82 KB
    unsigned short* hB = (unsigned short*)carve((size_t)2 * BSZ * KP * 2);    // 164 KB
    float*          cF = (float*)carve((size_t)2 * BSZ * NP * 4);             // 307 KB
    float*          z  = (float*)carve((size_t)2 * BSZ * G4 * 4);             // 1.2 MB

    // one-time prep (re-run every call: deterministic, replay-safe)
    k_zero_state<<<(2 * BSZ * KP + 255) / 256, 256, 0, stream>>>(cF, hB, out);
    k_conv_w<<<dim3((KP2 * G4 + 255) / 256, 2), 256, 0, stream>>>(W_fw, W_bw, Wp);
    k_conv_p<<<(KP2 * NC + 255) / 256, 256, 0, stream>>>(P, Pp);
    k_conv_e<<<(BSZ * TLEN * KP + 255) / 256, 256, 0, stream>>>(e, eB);

    // sequential recurrence: both directions per launch
    for (int s = 0; s < TLEN; ++s) {
        k_step_gemm<<<dim3(G4 / 16, 2), 256, 0, stream>>>(eB, hB, Wp, b_fw, b_bw, z, s);
        k_step_gate<<<(2 * BSZ * NP + 255) / 256, 256, 0, stream>>>(z, cF, hB, nlen, s);
    }

    k_final<<<1, 256, 0, stream>>>(hB, Pp, sT, out);
}
